// _Upsample1d_81655918231915
// MI455X (gfx1250) — matrix operand order repaired and checked by reading, not compiled, not run on hardware
//
#include <hip/hip_runtime.h>

typedef __attribute__((ext_vector_type(16))) __bf16 v16bf;
typedef __attribute__((ext_vector_type(8)))  __bf16 v8bf;
typedef __attribute__((ext_vector_type(8)))  float  v8f;

#define L_IN   2048
#define N_FULL 4108          // (2048 + 2*3) * 2
#define KP     2054          // nonzero (even) samples of zero-stuffed signal
#define M_ROWS 1024          // B*C
#define N_OUT  4096
#define PADC   3

#define TM 128
#define TN 64
#define TK 32

static __device__ __forceinline__ v16bf cat8(v8bf a, v8bf b) {
  return __builtin_shufflevector(a, b, 0,1,2,3,4,5,6,7,8,9,10,11,12,13,14,15);
}

static __device__ __forceinline__ void bsplit(float v, __bf16* hi, __bf16* lo) {
  __bf16 h = (__bf16)v;
  *hi = h;
  *lo = (__bf16)(v - (float)h);
}

// Ff[k] = sum_{m=0..7} f_padded[m] * dct_mat[m, k],  f_padded[m] = 2*up_filter[7-m]
__global__ void _up1d_ff_kernel(const float* __restrict__ up,
                                const float* __restrict__ dct,
                                float* __restrict__ Ff) {
  const int k = blockIdx.x * 256 + threadIdx.x;
  if (k >= N_FULL) return;
  float s = 0.f;
#pragma unroll
  for (int m = 0; m < 8; ++m)
    s += 2.0f * up[7 - m] * dct[(size_t)m * N_FULL + k];
  Ff[k] = s;
}

// GEMM-1: Xs[r,c] = Ff[c] * sum_i xp[r,i] * dct_mat[2i, c]   (reflect pad fused)
__global__ __launch_bounds__(256) void _up1d_gemm1(const float* __restrict__ x,
                                                   const float* __restrict__ dct,
                                                   const float* __restrict__ Ff,
                                                   float* __restrict__ Xs) {
  __shared__ __bf16 lAh[TM * TK], lAl[TM * TK];
  __shared__ __bf16 lBh[TN * TK], lBl[TN * TK];
  const int bm = blockIdx.x, bn = blockIdx.y;
  const int tid  = threadIdx.x;
  const int lane = tid & 31;
  const int wid  = tid >> 5;
  const int wm   = (wid & 3) * 32;
  const int wn   = (wid >> 2) * 32;
  const int hl   = (lane >> 4) & 1;    // lane half (0: lanes 0-15, 1: lanes 16-31)
  const int lm   = lane & 15;

  v8f acc[2][2] = {};

  const int nk = (KP + TK - 1) / TK;   // 65
  for (int kt = 0; kt < nk; ++kt) {
    const int k0 = kt * TK;
    // ---- stage A: 128x32 of reflect-padded x (even taps of zero-stuffed xu)
    for (int idx = tid; idx < TM * TK; idx += 256) {
      const int r = idx >> 5, kk = idx & 31;
      const int i = k0 + kk;
      float v = 0.f;
      if (i < KP) {
        int s = i - PADC;
        s = (s < 0) ? -s : s;
        s = (s > L_IN - 1) ? (2 * (L_IN - 1) - s) : s;
        v = x[(size_t)(bm * TM + r) * L_IN + s];
      }
      __bf16 h, l; bsplit(v, &h, &l);
      lAh[idx] = h; lAl[idx] = l;
    }
    // ---- stage B transposed: lB[n*TK+kk] = dct[(2*(k0+kk)) * N + bn*64 + n]
    for (int idx = tid; idx < TN * TK; idx += 256) {
      const int kk = idx >> 6, n = idx & 63;
      const int kr = k0 + kk;
      const int c  = bn * TN + n;
      float v = 0.f;
      if (kr < KP && c < N_FULL) {
        const float* p = &dct[(size_t)(2 * kr) * N_FULL + c];
        v = *p;
        if (kr + TK < KP) __builtin_prefetch(p + (size_t)(2 * TK) * N_FULL, 0, 0);
      }
      __bf16 h, l; bsplit(v, &h, &l);
      lBh[n * TK + kk] = h; lBl[n * TK + kk] = l;
    }
    __syncthreads();
#pragma unroll
    for (int i = 0; i < 2; ++i) {
      const int am = wm + i * 16 + lm;
      const v16bf ah = cat8(*(const v8bf*)&lAh[am * TK + hl * 8],
                            *(const v8bf*)&lAh[am * TK + 16 + hl * 8]);
      const v16bf al = cat8(*(const v8bf*)&lAl[am * TK + hl * 8],
                            *(const v8bf*)&lAl[am * TK + 16 + hl * 8]);
#pragma unroll
      for (int j = 0; j < 2; ++j) {
        const int bcol = wn + j * 16 + lm;
        const v16bf bh = cat8(*(const v8bf*)&lBh[bcol * TK + hl * 8],
                              *(const v8bf*)&lBh[bcol * TK + 16 + hl * 8]);
        const v16bf bl = cat8(*(const v8bf*)&lBl[bcol * TK + hl * 8],
                              *(const v8bf*)&lBl[bcol * TK + 16 + hl * 8]);
        acc[i][j] = __builtin_amdgcn_wmma_f32_16x16x32_bf16(false, ah, false, bh, (short)0, acc[i][j], false, false);
        acc[i][j] = __builtin_amdgcn_wmma_f32_16x16x32_bf16(false, ah, false, bl, (short)0, acc[i][j], false, false);
        acc[i][j] = __builtin_amdgcn_wmma_f32_16x16x32_bf16(false, al, false, bh, (short)0, acc[i][j], false, false);
      }
    }
    __syncthreads();
  }
  // ---- epilogue: scale by Ff, store f32 intermediate
#pragma unroll
  for (int i = 0; i < 2; ++i) {
#pragma unroll
    for (int j = 0; j < 2; ++j) {
      const int cb = bn * TN + wn + j * 16 + lm;
      if (cb < N_FULL) {
        const float fk = Ff[cb];
#pragma unroll
        for (int r = 0; r < 8; ++r) {
          const int m = bm * TM + wm + i * 16 + hl * 8 + r;
          Xs[(size_t)m * N_FULL + cb] = fk * acc[i][j][r];
        }
      }
    }
  }
}

// GEMM-2: y[r,j] = sum_k Xs[r,k] * idct_mat[k, j+6]   (crop fused)
__global__ __launch_bounds__(256) void _up1d_gemm2(const float* __restrict__ Xs,
                                                   const float* __restrict__ idct,
                                                   float* __restrict__ y) {
  __shared__ __bf16 lAh[TM * TK], lAl[TM * TK];
  __shared__ __bf16 lBh[TN * TK], lBl[TN * TK];
  const int bm = blockIdx.x, bn = blockIdx.y;
  const int tid  = threadIdx.x;
  const int lane = tid & 31;
  const int wid  = tid >> 5;
  const int wm   = (wid & 3) * 32;
  const int wn   = (wid >> 2) * 32;
  const int hl   = (lane >> 4) & 1;
  const int lm   = lane & 15;

  v8f acc[2][2] = {};

  const int nk = (N_FULL + TK - 1) / TK;  // 129
  for (int kt = 0; kt < nk; ++kt) {
    const int k0 = kt * TK;
    for (int idx = tid; idx < TM * TK; idx += 256) {
      const int r = idx >> 5, kk = idx & 31;
      const int k = k0 + kk;
      float v = (k < N_FULL) ? Xs[(size_t)(bm * TM + r) * N_FULL + k] : 0.f;
      __bf16 h, l; bsplit(v, &h, &l);
      lAh[idx] = h; lAl[idx] = l;
    }
    for (int idx = tid; idx < TN * TK; idx += 256) {
      const int kk = idx >> 6, n = idx & 63;
      const int k = k0 + kk;
      const int c = bn * TN + n;               // < 4096 always
      float v = 0.f;
      if (k < N_FULL) {
        const float* p = &idct[(size_t)k * N_FULL + 2 * PADC + c];
        v = *p;
        if (k + TK < N_FULL) __builtin_prefetch(p + (size_t)TK * N_FULL, 0, 0);
      }
      __bf16 h, l; bsplit(v, &h, &l);
      lBh[n * TK + kk] = h; lBl[n * TK + kk] = l;
    }
    __syncthreads();
#pragma unroll
    for (int i = 0; i < 2; ++i) {
      const int am = wm + i * 16 + lm;
      const v16bf ah = cat8(*(const v8bf*)&lAh[am * TK + hl * 8],
                            *(const v8bf*)&lAh[am * TK + 16 + hl * 8]);
      const v16bf al = cat8(*(const v8bf*)&lAl[am * TK + hl * 8],
                            *(const v8bf*)&lAl[am * TK + 16 + hl * 8]);
#pragma unroll
      for (int j = 0; j < 2; ++j) {
        const int bcol = wn + j * 16 + lm;
        const v16bf bh = cat8(*(const v8bf*)&lBh[bcol * TK + hl * 8],
                              *(const v8bf*)&lBh[bcol * TK + 16 + hl * 8]);
        const v16bf bl = cat8(*(const v8bf*)&lBl[bcol * TK + hl * 8],
                              *(const v8bf*)&lBl[bcol * TK + 16 + hl * 8]);
        acc[i][j] = __builtin_amdgcn_wmma_f32_16x16x32_bf16(false, ah, false, bh, (short)0, acc[i][j], false, false);
        acc[i][j] = __builtin_amdgcn_wmma_f32_16x16x32_bf16(false, ah, false, bl, (short)0, acc[i][j], false, false);
        acc[i][j] = __builtin_amdgcn_wmma_f32_16x16x32_bf16(false, al, false, bh, (short)0, acc[i][j], false, false);
      }
    }
    __syncthreads();
  }
#pragma unroll
  for (int i = 0; i < 2; ++i) {
#pragma unroll
    for (int j = 0; j < 2; ++j) {
      const int cb = bn * TN + wn + j * 16 + lm;   // < 4096 always
#pragma unroll
      for (int r = 0; r < 8; ++r) {
        const int m = bm * TM + wm + i * 16 + hl * 8 + r;
        y[(size_t)m * N_OUT + cb] = acc[i][j][r];
      }
    }
  }
}

extern "C" void kernel_launch(void* const* d_in, const int* in_sizes, int n_in,
                              void* d_out, int out_size, void* d_ws, size_t ws_size,
                              hipStream_t stream) {
  const float* x    = (const float*)d_in[0];   // [8,128,2048]
  const float* up   = (const float*)d_in[1];   // [15]
  const float* dct  = (const float*)d_in[2];   // [4108,4108]
  const float* idct = (const float*)d_in[3];   // [4108,4108]
  float* out = (float*)d_out;                  // [8,128,4096]

  float* Ff = (float*)d_ws;                           // 4108 f32
  float* Xs = (float*)((char*)d_ws + 16384);          // 1024 x 4108 f32

  _up1d_ff_kernel<<<(N_FULL + 255) / 256, 256, 0, stream>>>(up, dct, Ff);
  _up1d_gemm1<<<dim3(M_ROWS / TM, (N_FULL + TN - 1) / TN), 256, 0, stream>>>(x, dct, Ff, Xs);
  _up1d_gemm2<<<dim3(M_ROWS / TM, N_OUT / TN), 256, 0, stream>>>(Xs, idct, out);
}
